// Encoder_42099269435970
// MI455X (gfx1250) — compile-verified, matrix operand order checked by reading, not run on hardware
//
#include <hip/hip_runtime.h>

// ---------------------------------------------------------------------------
// Types for CDNA5 WMMA
// ---------------------------------------------------------------------------
typedef __bf16 bf16;
typedef bf16  v16bf __attribute__((ext_vector_type(16)));
typedef bf16  bf16x8 __attribute__((ext_vector_type(8)));
typedef float v8f   __attribute__((ext_vector_type(8)));

#define HID   512
#define G3    (3 * HID)
#define TPB   256
#define SLICE 32          // hidden units per recurrent workgroup (HID / 16 blocks)

// ---------------------------------------------------------------------------
// Init: deg = 2 (two self loops), h buffer(0) = hidden, barrier counter = 0
// ---------------------------------------------------------------------------
__global__ void init_kernel(float* __restrict__ deg, float* __restrict__ hbuf,
                            int* __restrict__ cnt, const float* __restrict__ hidden,
                            int n) {
  int i = blockIdx.x * blockDim.x + threadIdx.x;
  if (i < n)   deg[i] = 2.0f;
  if (i < HID) hbuf[i] = hidden[i];
  if (i == 0)  cnt[0] = 0;
}

__global__ void edge_deg_kernel(const int* __restrict__ col, float* __restrict__ deg, int E) {
  int e = blockIdx.x * blockDim.x + threadIdx.x;
  if (e < E) unsafeAtomicAdd(&deg[col[e]], 1.0f);
}

__global__ void dinv_kernel(const float* __restrict__ deg, float* __restrict__ dinv, int n) {
  int i = blockIdx.x * blockDim.x + threadIdx.x;
  if (i < n) dinv[i] = rsqrtf(deg[i]);   // deg >= 2 always
}

// ---------------------------------------------------------------------------
// LDS-staged bf16 WMMA GEMM:  C[M,N] = A[M,K] @ op(B) (+ bias)
//   transB==0 : B is [K,N] row-major
//   transB==1 : B is [N,K] row-major (we use B^T)
// Block tile 128x128, 8 waves, each wave 32x64 = 2x4 WMMA 16x16x32 fragments.
// Both A (m-major) and B (n-major) LDS tiles give contiguous 16B per-lane
// fragment runs -> ds_load_b128, no scalar LDS traffic in the hot loop.
// ---------------------------------------------------------------------------
#define BM 128
#define BN 128
#define BK 32
#define BSTRIDE (BK + 8)   // 40 bf16 = 80 B rows: 16B-aligned, bank-spread

__global__ void __launch_bounds__(TPB)
gemm_bf16_wmma(const float* __restrict__ A, const float* __restrict__ B,
               const float* __restrict__ bias, float* __restrict__ C,
               int M, int N, int K, int transB, int hasBias)
{
  __shared__ bf16 As[BM][BK];        // 8 KB, row (m) major
  __shared__ bf16 Bs[BN][BSTRIDE];   // 10 KB, col (n) major

  const int tid  = threadIdx.x;
  const int lane = tid & 31;
  const int wave = tid >> 5;
  const int wm   = wave & 3;             // 0..3  -> 32-row band
  const int wn   = wave >> 2;            // 0..1  -> 64-col band
  const int row0 = blockIdx.y * BM;
  const int col0 = blockIdx.x * BN;

  v8f acc[2][4] = {};

  const int ar = tid >> 1;               // 0..127
  const int ac = (tid & 1) * 16;         // 0 | 16

  for (int k0 = 0; k0 < K; k0 += BK) {
    // ---- stage A (128x32 fp32 -> bf16, m-major) ----
    {
      const float* ap = A + (size_t)(row0 + ar) * K + k0 + ac;
      bf16x8 pk[2];
#pragma unroll
      for (int h = 0; h < 2; ++h) {
        float4 v0 = ((const float4*)ap)[2 * h + 0];
        float4 v1 = ((const float4*)ap)[2 * h + 1];
        pk[h][0] = (bf16)v0.x; pk[h][1] = (bf16)v0.y;
        pk[h][2] = (bf16)v0.z; pk[h][3] = (bf16)v0.w;
        pk[h][4] = (bf16)v1.x; pk[h][5] = (bf16)v1.y;
        pk[h][6] = (bf16)v1.z; pk[h][7] = (bf16)v1.w;
        *(bf16x8*)(&As[ar][ac + 8 * h]) = pk[h];
      }
    }
    // ---- stage B tile (32 x 128), stored n-major Bs[n][k] ----
    if (transB == 0) {
      const int bk = tid >> 3;             // 0..31
      const int nb = (tid & 7) * 16;       // 0..112
      const float* bp = B + (size_t)(k0 + bk) * N + col0 + nb;
#pragma unroll
      for (int q = 0; q < 4; ++q) {
        float4 v = ((const float4*)bp)[q];
        Bs[nb + 4 * q + 0][bk] = (bf16)v.x;
        Bs[nb + 4 * q + 1][bk] = (bf16)v.y;
        Bs[nb + 4 * q + 2][bk] = (bf16)v.z;
        Bs[nb + 4 * q + 3][bk] = (bf16)v.w;
      }
    } else {
      const int bn   = tid >> 1;           // 0..127
      const int kseg = (tid & 1) * 16;     // 0 | 16
      const float* bp = B + (size_t)(col0 + bn) * K + k0 + kseg;
      bf16x8 pk[2];
#pragma unroll
      for (int h = 0; h < 2; ++h) {
        float4 v0 = ((const float4*)bp)[2 * h + 0];
        float4 v1 = ((const float4*)bp)[2 * h + 1];
        pk[h][0] = (bf16)v0.x; pk[h][1] = (bf16)v0.y;
        pk[h][2] = (bf16)v0.z; pk[h][3] = (bf16)v0.w;
        pk[h][4] = (bf16)v1.x; pk[h][5] = (bf16)v1.y;
        pk[h][6] = (bf16)v1.z; pk[h][7] = (bf16)v1.w;
        *(bf16x8*)(&Bs[bn][kseg + 8 * h]) = pk[h];
      }
    }
    __syncthreads();

    // ---- fragments per CDNA5 16-bit layout: two contiguous 8-elem runs ----
    const int mlane = lane & 15;
    const int kb    = (lane >> 4) * 8;   // 0 | 8
    v16bf afrag[2], bfrag[4];
#pragma unroll
    for (int tm = 0; tm < 2; ++tm) {
      const bf16* apf = &As[wm * 32 + tm * 16 + mlane][0];
      bf16x8 lo = *(const bf16x8*)(apf + kb);
      bf16x8 hi = *(const bf16x8*)(apf + kb + 16);
      afrag[tm] = __builtin_shufflevector(lo, hi, 0, 1, 2, 3, 4, 5, 6, 7,
                                          8, 9, 10, 11, 12, 13, 14, 15);
    }
#pragma unroll
    for (int tn = 0; tn < 4; ++tn) {
      const bf16* bpf = &Bs[wn * 64 + tn * 16 + mlane][0];
      bf16x8 lo = *(const bf16x8*)(bpf + kb);
      bf16x8 hi = *(const bf16x8*)(bpf + kb + 16);
      bfrag[tn] = __builtin_shufflevector(lo, hi, 0, 1, 2, 3, 4, 5, 6, 7,
                                          8, 9, 10, 11, 12, 13, 14, 15);
    }
#pragma unroll
    for (int tm = 0; tm < 2; ++tm)
#pragma unroll
      for (int tn = 0; tn < 4; ++tn)
        acc[tm][tn] = __builtin_amdgcn_wmma_f32_16x16x32_bf16(
            false, afrag[tm], false, bfrag[tn], (short)0, acc[tm][tn], false, false);

    __syncthreads();
  }

  // ---- epilogue: C/D layout (lanes 0-15: M=r, lanes 16-31: M=8+r) ----
  const int nlane = lane & 15;
  const int moff  = (lane >> 4) * 8;
#pragma unroll
  for (int tm = 0; tm < 2; ++tm) {
#pragma unroll
    for (int tn = 0; tn < 4; ++tn) {
      const int gc = col0 + wn * 64 + tn * 16 + nlane;
      const float bv = hasBias ? bias[gc] : 0.0f;
#pragma unroll
      for (int r = 0; r < 8; ++r) {
        const int gr = row0 + wm * 32 + tm * 16 + moff + r;
        C[(size_t)gr * N + gc] = acc[tm][tn][r] + bv;
      }
    }
  }
}

// ---------------------------------------------------------------------------
// GCN aggregation: self-loop term (plain store) then edge scatter (hw atomics)
// ---------------------------------------------------------------------------
__global__ void gcn_self_kernel(const float* __restrict__ xw, const float* __restrict__ dinv,
                                const float* __restrict__ bias, float* __restrict__ X) {
  int idx = blockIdx.x * blockDim.x + threadIdx.x;
  int i = idx >> 9;          // / HID
  int c = idx & (HID - 1);
  float d = dinv[i];
  X[idx] = 2.0f * d * d * xw[idx] + bias[c];
}

__global__ void gcn_edge_kernel(const int* __restrict__ row, const int* __restrict__ col,
                                const float* __restrict__ dinv, const float* __restrict__ xw,
                                float* __restrict__ X) {
  const int e = blockIdx.x;
  const int r = row[e];
  const int c = col[e];
  const float nrm = dinv[r] * dinv[c];
  const float* src = xw + (size_t)r * HID;
  float* dst = X + (size_t)c * HID;
  for (int ch = threadIdx.x; ch < HID; ch += blockDim.x)
    unsafeAtomicAdd(&dst[ch], nrm * src[ch]);
}

// ---------------------------------------------------------------------------
// Recurrent GRU: 16 persistent blocks, w_hh slice resident in 192 KB LDS
// (CDNA5 320 KB/WGP). Weights preloaded with the CDNA5 async DMA engine
// (global_load_async_to_lds_b128 / ASYNCcnt). Double-buffered h -> one grid
// barrier per step.
// ---------------------------------------------------------------------------
__global__ void __launch_bounds__(TPB)
gru_kernel(const float* __restrict__ gi_all,   // [T, 3H] precomputed w_ih@x_t + b_ih
           const float* __restrict__ w_hh,     // [3H, H]
           const float* __restrict__ b_hh,     // [3H]
           float* __restrict__ hbuf,           // [2, H] double buffer
           int* __restrict__ cnt,              // grid barrier counter (monotonic)
           float* __restrict__ out,            // [T*H] outs, then [H] h_last
           int T)
{
  extern __shared__ float sm[];
  float* Ws  = sm;                       // [3*SLICE][HID] = 192 KB
  float* h_s = Ws + 3 * SLICE * HID;     // [HID]

  const int tid = threadIdx.x;
  const int j0  = blockIdx.x * SLICE;
  const int nb  = gridDim.x;

  // Async-DMA this block's 96 rows of w_hh (3 contiguous 64 KB chunks) to LDS.
  {
    const int gate_bytes = SLICE * HID * 4;   // 65536
#pragma unroll
    for (int g = 0; g < 3; ++g) {
      const char* src = (const char*)(w_hh + (size_t)(g * HID + j0) * HID);
      char*       dst = (char*)(Ws + g * SLICE * HID);
      for (int ofs = tid * 16; ofs < gate_bytes; ofs += TPB * 16) {
        unsigned int       lds_addr = (unsigned int)(size_t)(dst + ofs);
        unsigned long long gaddr    = (unsigned long long)(size_t)(src + ofs);
        asm volatile("global_load_async_to_lds_b128 %0, %1, off"
                     :: "v"(lds_addr), "v"(gaddr) : "memory");
      }
    }
    asm volatile("s_wait_asynccnt 0" ::: "memory");
  }
  __syncthreads();

  const int sub   = tid & 7;             // 8 lanes cooperate per row
  const int rloc  = tid >> 3;            // 0..31 row within slice
  const int cbase = sub * (HID / 8);     // 64-element segment
  const int jg    = j0 + rloc;

  const float bh_r = b_hh[jg];
  const float bh_z = b_hh[HID + jg];
  const float bh_n = b_hh[2 * HID + jg];

  for (int t = 0; t < T; ++t) {
    const float* hin  = hbuf + (t & 1) * HID;
    float*       hout = hbuf + ((t + 1) & 1) * HID;
    const float* gi_t = gi_all + (size_t)t * G3;

    __builtin_prefetch(gi_t + G3 + (tid << 2), 0, 1);   // next step's gates

    for (int i = tid; i < HID; i += TPB) h_s[i] = hin[i];
    __syncthreads();

    // fused r/z/n partial dot products over this thread's 64-element segment
    float pr = 0.f, pz = 0.f, pn = 0.f;
    const float* wr = Ws + (size_t)rloc * HID + cbase;
    const float* wz = Ws + (size_t)(SLICE + rloc) * HID + cbase;
    const float* wn = Ws + (size_t)(2 * SLICE + rloc) * HID + cbase;
    const float* hv = h_s + cbase;
#pragma unroll 16
    for (int c = 0; c < HID / 8; ++c) {
      const float h = hv[c];
      pr = fmaf(wr[c], h, pr);
      pz = fmaf(wz[c], h, pz);
      pn = fmaf(wn[c], h, pn);
    }
    // width-8 shuffle reduction (wave32)
    pr += __shfl_down(pr, 4, 8); pr += __shfl_down(pr, 2, 8); pr += __shfl_down(pr, 1, 8);
    pz += __shfl_down(pz, 4, 8); pz += __shfl_down(pz, 2, 8); pz += __shfl_down(pz, 1, 8);
    pn += __shfl_down(pn, 4, 8); pn += __shfl_down(pn, 2, 8); pn += __shfl_down(pn, 1, 8);

    if (sub == 0) {
      const float ghr = pr + bh_r, ghz = pz + bh_z, ghn = pn + bh_n;
      const float rv = 1.0f / (1.0f + __expf(-(gi_t[jg] + ghr)));
      const float zv = 1.0f / (1.0f + __expf(-(gi_t[HID + jg] + ghz)));
      const float ng = tanhf(gi_t[2 * HID + jg] + rv * ghn);
      const float hn = (1.0f - zv) * ng + zv * h_s[jg];
      out[(size_t)t * HID + jg] = hn;
      hout[jg] = hn;
      if (t == T - 1) out[(size_t)T * HID + jg] = hn;   // h_last tail
    }
    __threadfence();
    __syncthreads();
    if (tid == 0) {
      atomicAdd(cnt, 1);
      const int target = (t + 1) * nb;
      while (atomicAdd(cnt, 0) < target) __builtin_amdgcn_s_sleep(1);
    }
    __syncthreads();
    __threadfence();
  }
}

// ---------------------------------------------------------------------------
// Host launcher
// ---------------------------------------------------------------------------
extern "C" void kernel_launch(void* const* d_in, const int* in_sizes, int n_in,
                              void* d_out, int out_size, void* d_ws, size_t ws_size,
                              hipStream_t stream) {
  const float* x      = (const float*)d_in[0];   // [N, 512]
  const int*   eidx   = (const int*)  d_in[1];   // [2, E]
  const float* hidden = (const float*)d_in[2];   // [1, 512]
  const float* gw     = (const float*)d_in[3];   // [512, 512]
  const float* gb     = (const float*)d_in[4];   // [512]
  const float* w_ih   = (const float*)d_in[5];   // [1536, 512]
  const float* w_hh   = (const float*)d_in[6];   // [1536, 512]
  const float* b_ih   = (const float*)d_in[7];   // [1536]
  const float* b_hh   = (const float*)d_in[8];   // [1536]
  float* out = (float*)d_out;

  const int N = in_sizes[0] / HID;   // 16384
  const int E = in_sizes[1] / 2;     // 262144

  // workspace layout (floats)
  float* XW   = (float*)d_ws;                    // [N, 512]
  float* X    = XW + (size_t)N * HID;            // [N, 512] GCN output
  float* GI   = X + (size_t)N * HID;             // [N, 1536] precomputed input gates
  float* deg  = GI + (size_t)N * G3;             // [N]
  float* dinv = deg + N;                         // [N]
  float* hbuf = dinv + N;                        // [2, 512]
  int*   cnt  = (int*)(hbuf + 2 * HID);          // [1]

  const int* erow = eidx;       // sources
  const int* ecol = eidx + E;   // targets

  init_kernel<<<(N + TPB - 1) / TPB, TPB, 0, stream>>>(deg, hbuf, cnt, hidden, N);
  edge_deg_kernel<<<(E + TPB - 1) / TPB, TPB, 0, stream>>>(ecol, deg, E);
  dinv_kernel<<<(N + TPB - 1) / TPB, TPB, 0, stream>>>(deg, dinv, N);

  // XW = x @ W  (bf16 WMMA)
  gemm_bf16_wmma<<<dim3(HID / BN, N / BM), TPB, 0, stream>>>(
      x, gw, nullptr, XW, N, HID, HID, /*transB=*/0, /*hasBias=*/0);

  // GCN aggregation
  gcn_self_kernel<<<(size_t)N * HID / TPB, TPB, 0, stream>>>(XW, dinv, gb, X);
  gcn_edge_kernel<<<E, TPB, 0, stream>>>(erow, ecol, dinv, XW, X);

  // GI = X @ w_ih^T + b_ih  (bf16 WMMA) — hoists all sequential-independent GRU work
  gemm_bf16_wmma<<<dim3(G3 / BN, N / BM), TPB, 0, stream>>>(
      X, w_ih, b_ih, GI, N, G3, HID, /*transB=*/1, /*hasBias=*/1);

  // Sequential GRU chain: 16 persistent blocks, LDS-resident w_hh slices
  const size_t smem = (size_t)(3 * SLICE * HID + HID) * sizeof(float);  // ~194 KB
  gru_kernel<<<16, TPB, smem, stream>>>(GI, w_hh, b_hh, hbuf, cnt, out, N);
}
